// UltraProfessionalASIAttention_80152679678419
// MI455X (gfx1250) — matrix operand order repaired and checked by reading, not compiled, not run on hardware
//
#include <hip/hip_runtime.h>
#include <hip/hip_bf16.h>
#include <stdint.h>

// ---------------------------------------------------------------------------
// Linear-attention block for MI455X (gfx1250, wave32).
// All GEMMs: v_wmma_f32_16x16x32_bf16, double-buffered LDS tiles staged by the
// Tensor Data Mover (tensor_load_to_lds + s_wait_tensorcnt). B operands are
// kept transposed [N,K] in memory so both A and B fragments are vectorized
// ds_load_b128 reads from TDM-padded LDS rows.
// ---------------------------------------------------------------------------

typedef __attribute__((ext_vector_type(16))) __bf16 v16bf;
typedef __attribute__((ext_vector_type(8)))  float  v8f;
typedef __attribute__((ext_vector_type(4)))  unsigned int v4u;
typedef __attribute__((ext_vector_type(4)))  int    v4i;
typedef __attribute__((ext_vector_type(8)))  int    v8i;

__device__ __forceinline__ unsigned short f2bf(float f) {
    union { float f; uint32_t u; } x; x.f = f;
    uint32_t u = x.u;
    return (unsigned short)((u + 0x7FFFu + ((u >> 16) & 1u)) >> 16);
}
__device__ __forceinline__ float bf2f(unsigned short s) {
    union { uint32_t u; float f; } x; x.u = ((uint32_t)s) << 16;
    return x.f;
}
__device__ __forceinline__ uint32_t rfl(uint32_t v) {
    return (uint32_t)__builtin_amdgcn_readfirstlane((int)v);
}

// ------------------- Tensor Data Mover: 2D tile -> LDS ---------------------
__device__ __forceinline__ void tdm_load_2d(uint32_t lds_addr,
                                            const unsigned short* gsrc,
                                            uint32_t td0, uint32_t td1,
                                            uint32_t stride,
                                            uint32_t pint, uint32_t pamt)
{
    uint64_t ga = (uint64_t)(uintptr_t)gsrc;
    uint32_t g0w0 = rfl(1u);                                    // count=1
    uint32_t g0w1 = rfl(lds_addr);
    uint32_t g0w2 = rfl((uint32_t)(ga & 0xffffffffu));
    uint32_t g0w3 = rfl((uint32_t)((ga >> 32) & 0x01ffffffu) | (2u << 30));

    uint32_t g1w0 = rfl((1u << 16) | (1u << 20) | (pint << 22) | (pamt << 25));
    uint32_t g1w1 = rfl((td0 & 0xffffu) << 16);
    uint32_t g1w2 = rfl((td0 >> 16) | ((td1 & 0xffffu) << 16));
    uint32_t g1w3 = rfl((td1 >> 16) | ((td0 & 0xffffu) << 16));
    uint32_t g1w4 = rfl(td1 & 0xffffu);
    uint32_t g1w5 = rfl(stride);
    uint32_t g1w6 = rfl(0u);
    uint32_t g1w7 = rfl(0u);

    v4u g0 = { g0w0, g0w1, g0w2, g0w3 };
    v8i g1 = { (int)g1w0, (int)g1w1, (int)g1w2, (int)g1w3,
               (int)g1w4, (int)g1w5, (int)g1w6, (int)g1w7 };
    v4i gz = { 0, 0, 0, 0 };
#if __clang_major__ >= 23
    v8i gz8 = { 0, 0, 0, 0, 0, 0, 0, 0 };
    __builtin_amdgcn_tensor_load_to_lds(g0, g1, gz, gz, gz8, 0);
#else
    __builtin_amdgcn_tensor_load_to_lds(g0, g1, gz, gz, 0);
#endif
}

// -------------------------------- converts --------------------------------
__global__ void cvt_f32_bf16(const float* __restrict__ src,
                             unsigned short* __restrict__ dst, int n) {
    int i = blockIdx.x * blockDim.x + threadIdx.x;
    int stride = gridDim.x * blockDim.x;
    for (; i < n; i += stride) dst[i] = f2bf(src[i]);
}

// f32 [R,C] -> bf16 transposed [C,R] via 32x32 LDS tile
__global__ __launch_bounds__(256)
void cvt_f32_bf16_T(const float* __restrict__ src,
                    unsigned short* __restrict__ dst, int R, int C) {
    __shared__ float tile[32][33];
    int bx = blockIdx.x * 32;               // C
    int by = blockIdx.y * 32;               // R
    int tx = threadIdx.x & 31, ty = threadIdx.x >> 5;
    #pragma unroll
    for (int i = 0; i < 32; i += 8)
        tile[ty + i][tx] = src[(long long)(by + ty + i) * C + bx + tx];
    __syncthreads();
    #pragma unroll
    for (int i = 0; i < 32; i += 8)
        dst[(long long)(bx + ty + i) * R + by + tx] = f2bf(tile[tx][ty + i]);
}

// ----------------------------- WMMA GEMM (TDM-fed) -------------------------
#define BM 128
#define BN 128
#define BK 32
#define TS  40       // padded LDS row stride (ushorts): TDM pint=3, pamt=3
#define BUF_USH 5120 // 128 * 40

enum { OUT_BF16 = 0, OUT_F32 = 1, OUT_QKV = 2, OUT_QKVT = 3,
       OUT_BF16T = 4, OUT_CTX = 5 };

template <int MODE>
__global__ __launch_bounds__(256)
void gemm_tdm(const unsigned short* __restrict__ A,   // [M,K] row-major
              const unsigned short* __restrict__ Bt,  // [N,K] row-major (B^T)
              const float* __restrict__ bias,
              void* __restrict__ Cout,
              const float* __restrict__ denom,
              int K, int lda, int ldb, int ldc,
              long long strideA, long long strideB, long long strideC,
              int relu)
{
    __shared__ __align__(16) unsigned short smem[4 * BUF_USH];

    const int tid  = threadIdx.x;
    const int lane = tid & 31;
    const int wv   = tid >> 5;
    const int hf   = lane >> 4;
    const int l16  = lane & 15;
    const int wm   = wv >> 1;      // 0..3 : 32-row strip
    const int wn   = wv & 1;       // 0..1 : 64-col strip
    const int z    = blockIdx.z;
    const int m0   = blockIdx.y * BM;
    const int n0   = blockIdx.x * BN;

    const unsigned short* Ab = A  + (long long)z * strideA;
    const unsigned short* Bb = Bt + (long long)z * strideB;
    const uint32_t lds_base = (uint32_t)(uintptr_t)(void*)smem;

    v8f acc[2][4];
    #pragma unroll
    for (int i = 0; i < 2; i++)
        #pragma unroll
        for (int j = 0; j < 4; j++)
            #pragma unroll
            for (int r = 0; r < 8; r++) acc[i][j][r] = 0.0f;

    const int nT = K / BK;

    auto issue = [&](int buf, int k0) {
        uint32_t aoff = lds_base + (uint32_t)(buf * BUF_USH * 2);
        uint32_t boff = lds_base + (uint32_t)((2 + buf) * BUF_USH * 2);
        // A tile: 128 rows x 32 contiguous k
        tdm_load_2d(aoff, Ab + (long long)m0 * lda + k0, BK, BM,
                    (uint32_t)lda, 3u, 3u);
        // B^T tile: 128 rows (n) x 32 contiguous k
        tdm_load_2d(boff, Bb + (long long)n0 * ldb + k0, BK, BN,
                    (uint32_t)ldb, 3u, 3u);
    };

    if (wv == 0) {
        issue(0, 0);
        __builtin_amdgcn_s_wait_tensorcnt(0);
    }
    __syncthreads();

    for (int t = 0; t < nT; t++) {
        const int cur = t & 1;
        if (wv == 0 && (t + 1) < nT) issue(cur ^ 1, (t + 1) * BK);

        const unsigned short* As = smem + cur * BUF_USH;
        const unsigned short* Bs = smem + (2 + cur) * BUF_USH;

        union FragU { uint4 u[2]; v16bf v; };
        FragU afr[2], bfr[4];

        #pragma unroll
        for (int mi = 0; mi < 2; mi++) {
            int m = wm * 32 + mi * 16 + l16;
            // a[e] = A[m][(e>>3)*16 + hf*8 + (e&7)]
            afr[mi].u[0] = *(const uint4*)(As + m * TS + hf * 8);
            afr[mi].u[1] = *(const uint4*)(As + m * TS + 16 + hf * 8);
        }
        #pragma unroll
        for (int ni = 0; ni < 4; ni++) {
            int n = wn * 64 + ni * 16 + l16;
            // b[e] = B[(e>>3)*16 + hf*8 + (e&7)][n] = Bt[n][(e>>3)*16 + hf*8 + (e&7)]
            bfr[ni].u[0] = *(const uint4*)(Bs + n * TS + hf * 8);
            bfr[ni].u[1] = *(const uint4*)(Bs + n * TS + 16 + hf * 8);
        }

        #pragma unroll
        for (int mi = 0; mi < 2; mi++)
            #pragma unroll
            for (int ni = 0; ni < 4; ni++)
                acc[mi][ni] = __builtin_amdgcn_wmma_f32_16x16x32_bf16(
                    false, afr[mi].v, false, bfr[ni].v,
                    (short)0, acc[mi][ni], false, false);

        if (wv == 0 && (t + 1) < nT) __builtin_amdgcn_s_wait_tensorcnt(0);
        __syncthreads();
    }

    // ---- epilogue: c[r] -> (m = hf*8 + r, n = lane&15) ----
    const bool has_bias = (bias != nullptr);
    #pragma unroll
    for (int mi = 0; mi < 2; mi++) {
        const int gmb = m0 + wm * 32 + mi * 16 + hf * 8;  // 8-aligned row base
        float dsc[8];
        if (MODE == OUT_CTX) {
            #pragma unroll
            for (int r = 0; r < 8; r++)
                dsc[r] = 1.0f / (denom[(long long)z * 4096 + gmb + r] + 1e-8f);
        }
        #pragma unroll
        for (int ni = 0; ni < 4; ni++) {
            const int gn = n0 + wn * 64 + ni * 16 + l16;
            const float bval = has_bias ? bias[gn] : 0.0f;

            if (MODE == OUT_QKVT || MODE == OUT_BF16T) {
                // 8 consecutive-m values -> one 16B store
                unsigned short pk[8];
                #pragma unroll
                for (int r = 0; r < 8; r++) {
                    float v = acc[mi][ni][r] + bval;
                    if (relu) v = v > 0.0f ? v : 0.0f;
                    pk[r] = f2bf(v);
                }
                long long idx;
                if (MODE == OUT_QKVT) {
                    // v^T: [B,NH,HD,L]
                    int bidx = gmb >> 12, l = gmb & 4095;
                    int head = gn >> 7,   d = gn & 127;
                    idx = ((long long)(bidx * 16 + head)) * 524288 +
                          (long long)d * 4096 + l;
                } else {
                    idx = (long long)z * strideC + (long long)gn * ldc + gmb;
                }
                *(uint4*)&((unsigned short*)Cout)[idx] = *(uint4*)pk;
            } else {
                #pragma unroll
                for (int r = 0; r < 8; r++) {
                    int gm = gmb + r;
                    float v = acc[mi][ni][r] + bval;
                    if (relu) v = v > 0.0f ? v : 0.0f;
                    if (MODE == OUT_F32) {
                        ((float*)Cout)[(long long)z * strideC +
                                       (long long)gm * ldc + gn] = v;
                    } else if (MODE == OUT_BF16) {
                        ((unsigned short*)Cout)[(long long)z * strideC +
                                                (long long)gm * ldc + gn] = f2bf(v);
                    } else if (MODE == OUT_QKV) {
                        int bidx = gm >> 12, l = gm & 4095;
                        int head = gn >> 7,  d = gn & 127;
                        ((unsigned short*)Cout)[(((long long)(bidx * 16 + head)) << 19) +
                                                ((long long)l << 7) + d] = f2bf(v);
                    } else { // OUT_CTX
                        float s = v * dsc[r];
                        int bz = z >> 4, hz = z & 15;
                        ((unsigned short*)Cout)[(long long)bz * 4096 * 2048 +
                                                (long long)gm * 2048 +
                                                hz * 128 + gn] = f2bf(s);
                    }
                }
            }
        }
    }
}

// ----------------------- LayerNorm (one wave32 per row) --------------------
// In-place over rows of 256; optionally emits transposed copy (for kf^T).
__global__ __launch_bounds__(256)
void ln_rows_kernel(unsigned short* __restrict__ X,
                    const float* __restrict__ g, const float* __restrict__ b,
                    const float* __restrict__ mask, int kmode,
                    unsigned short* __restrict__ outT)
{
    int wv = threadIdx.x >> 5, lane = threadIdx.x & 31;
    long long row = (long long)blockIdx.x * 8 + wv;
    unsigned short* p = X + row * 256 + lane * 8;
    unsigned short buf[8];
    *(uint4*)buf = *(const uint4*)p;
    float v[8], s = 0.f, s2 = 0.f;
    #pragma unroll
    for (int i = 0; i < 8; i++) { v[i] = bf2f(buf[i]); s += v[i]; s2 += v[i] * v[i]; }
    #pragma unroll
    for (int off = 16; off > 0; off >>= 1) {
        s  += __shfl_xor(s,  off, 32);
        s2 += __shfl_xor(s2, off, 32);
    }
    float mu   = s * (1.0f / 256.0f);
    float var  = s2 * (1.0f / 256.0f) - mu * mu;
    float rstd = rsqrtf(var + 1e-5f);
    float fac = 1.0f;
    if (kmode) {
        long long bb = row >> 16;
        long long l  = row & 4095;
        fac = 1.0f + mask[bb * 4096 + l];
    }
    #pragma unroll
    for (int i = 0; i < 8; i++) {
        int j = lane * 8 + i;
        buf[i] = f2bf(((v[i] - mu) * rstd * g[j] + b[j]) * fac);
    }
    *(uint4*)p = *(uint4*)buf;
    if (outT) {
        long long bh = row >> 12, l = row & 4095;
        #pragma unroll
        for (int i = 0; i < 8; i++) {
            int j = lane * 8 + i;
            outT[bh * (256LL * 4096) + (long long)j * 4096 + l] = buf[i];
        }
    }
}

// ------------- k_sum over L (reads kf^T: contiguous per (bh,f)) ------------
__global__ __launch_bounds__(256)
void ksum_kernel(const unsigned short* __restrict__ kfT, float* __restrict__ ksum)
{
    int wv = threadIdx.x >> 5, lane = threadIdx.x & 31;
    long long row = (long long)blockIdx.x * 8 + wv;     // bh*256 + f
    const unsigned short* p = kfT + row * 4096;
    float acc = 0.0f;
    for (int it = 0; it < 16; it++) {
        unsigned short buf[8];
        *(uint4*)buf = *(const uint4*)(p + ((long long)it * 32 + lane) * 8);
        #pragma unroll
        for (int i = 0; i < 8; i++) acc += bf2f(buf[i]);
    }
    #pragma unroll
    for (int off = 16; off > 0; off >>= 1) acc += __shfl_xor(acc, off, 32);
    if (lane == 0) ksum[row] = acc;
}

// --------------------- denom: q_feat . k_sum per token ---------------------
__global__ __launch_bounds__(256)
void denom_kernel(const unsigned short* __restrict__ qf,
                  const float* __restrict__ ksum, float* __restrict__ denom)
{
    int wv = threadIdx.x >> 5, lane = threadIdx.x & 31;
    long long row = (long long)blockIdx.x * 8 + wv;
    int bh = (int)(row >> 12);
    unsigned short buf[8];
    *(uint4*)buf = *(const uint4*)(qf + row * 256 + lane * 8);
    const float* ks = ksum + bh * 256 + lane * 8;
    float s = 0.0f;
    #pragma unroll
    for (int i = 0; i < 8; i++) s += bf2f(buf[i]) * ks[i];
    #pragma unroll
    for (int off = 16; off > 0; off >>= 1) s += __shfl_xor(s, off, 32);
    if (lane == 0) denom[row] = s;
}

// ------------------------------- launcher ----------------------------------
extern "C" void kernel_launch(void* const* d_in, const int* in_sizes, int n_in,
                              void* d_out, int out_size, void* d_ws, size_t ws_size,
                              hipStream_t stream)
{
    (void)in_sizes; (void)n_in; (void)out_size; (void)ws_size;

    const float* X    = (const float*)d_in[0];
    const float* mask = (const float*)d_in[1];
    const float* Wq   = (const float*)d_in[2];
    const float* bq   = (const float*)d_in[3];
    const float* Wk   = (const float*)d_in[4];
    const float* bk   = (const float*)d_in[5];
    const float* Wv   = (const float*)d_in[6];
    const float* bv   = (const float*)d_in[7];
    const float* Wo   = (const float*)d_in[8];
    const float* bo   = (const float*)d_in[9];
    const float* Wf1  = (const float*)d_in[10];
    const float* bf1  = (const float*)d_in[11];
    const float* Wf2  = (const float*)d_in[12];
    const float* bf2  = (const float*)d_in[13];
    const float* lng  = (const float*)d_in[14];
    const float* lnb  = (const float*)d_in[15];

    char* ws = (char*)d_ws;
    size_t off = 0;
    auto alloc = [&](size_t bytes) -> char* {
        char* p = ws + off;
        off += (bytes + 255) & ~(size_t)255;
        return p;
    };

    unsigned short* Xb   = (unsigned short*)alloc(8192LL * 2048 * 2);
    unsigned short* WqT  = (unsigned short*)alloc(2048LL * 2048 * 2);
    unsigned short* WkT  = (unsigned short*)alloc(2048LL * 2048 * 2);
    unsigned short* WvT  = (unsigned short*)alloc(2048LL * 2048 * 2);
    unsigned short* WoT  = (unsigned short*)alloc(2048LL * 2048 * 2);
    unsigned short* Wf1T = (unsigned short*)alloc(256LL * 128 * 2);
    unsigned short* Wf2T = (unsigned short*)alloc(256LL * 256 * 2);
    unsigned short* qb   = (unsigned short*)alloc(131072LL * 128 * 2);
    unsigned short* kb   = (unsigned short*)alloc(131072LL * 128 * 2);
    unsigned short* vT   = (unsigned short*)alloc(32LL * 128 * 4096 * 2);
    unsigned short* h1   = (unsigned short*)alloc(131072LL * 256 * 2);
    unsigned short* qf   = (unsigned short*)alloc(131072LL * 256 * 2);
    unsigned short* kf   = (unsigned short*)alloc(131072LL * 256 * 2);
    unsigned short* kfT  = (unsigned short*)alloc(32LL * 256 * 4096 * 2);
    unsigned short* kvT  = (unsigned short*)alloc(32LL * 128 * 256 * 2);
    float*          ksum = (float*)alloc(32LL * 256 * 4);
    float*          dnm  = (float*)alloc(131072LL * 4);
    unsigned short* ctx  = (unsigned short*)alloc(8192LL * 2048 * 2);

    dim3 blk(256);
    auto cvt = [&](const float* s, unsigned short* d, int n) {
        cvt_f32_bf16<<<dim3((n + 255) / 256), blk, 0, stream>>>(s, d, n);
    };
    auto cvtT = [&](const float* s, unsigned short* d, int R, int C) {
        cvt_f32_bf16_T<<<dim3(C / 32, R / 32), blk, 0, stream>>>(s, d, R, C);
    };

    // 0) convert (weights transposed so GEMMs consume B^T [N,K])
    cvt (X,   Xb,   16777216);
    cvtT(Wq,  WqT,  2048, 2048);
    cvtT(Wk,  WkT,  2048, 2048);
    cvtT(Wv,  WvT,  2048, 2048);
    cvtT(Wo,  WoT,  2048, 2048);
    cvtT(Wf1, Wf1T, 128, 256);
    cvtT(Wf2, Wf2T, 256, 256);

    // 1) QKV projections -> q,k natural [B,NH,L,HD]; v transposed [B,NH,HD,L]
    {
        dim3 g(2048 / BN, 8192 / BM, 1);
        gemm_tdm<OUT_QKV ><<<g, blk, 0, stream>>>(Xb, WqT, bq, qb, nullptr,
            2048, 2048, 2048, 0, 0, 0, 0, 0);
        gemm_tdm<OUT_QKV ><<<g, blk, 0, stream>>>(Xb, WkT, bk, kb, nullptr,
            2048, 2048, 2048, 0, 0, 0, 0, 0);
        gemm_tdm<OUT_QKVT><<<g, blk, 0, stream>>>(Xb, WvT, bv, vT, nullptr,
            2048, 2048, 2048, 0, 0, 0, 0, 0);
    }

    // 2) feature maps: relu(x@Wf1+b1)@Wf2+b2, LayerNorm in place
    {
        dim3 g1(256 / BN, 131072 / BM, 1);
        gemm_tdm<OUT_BF16><<<g1, blk, 0, stream>>>(qb, Wf1T, bf1, h1, nullptr,
            128, 128, 128, 256, 0, 0, 0, 1);
        gemm_tdm<OUT_BF16><<<g1, blk, 0, stream>>>(h1, Wf2T, bf2, qf, nullptr,
            256, 256, 256, 256, 0, 0, 0, 0);
        ln_rows_kernel<<<dim3(16384), blk, 0, stream>>>(qf, lng, lnb, mask, 0,
                                                        nullptr);
        gemm_tdm<OUT_BF16><<<g1, blk, 0, stream>>>(kb, Wf1T, bf1, h1, nullptr,
            128, 128, 128, 256, 0, 0, 0, 1);
        gemm_tdm<OUT_BF16><<<g1, blk, 0, stream>>>(h1, Wf2T, bf2, kf, nullptr,
            256, 256, 256, 256, 0, 0, 0, 0);
        // also emits kf^T [B,NH,FD,L] for the kv GEMM
        ln_rows_kernel<<<dim3(16384), blk, 0, stream>>>(kf, lng, lnb, mask, 1,
                                                        kfT);
    }

    // 3) k_sum over L (contiguous rows of kf^T)
    ksum_kernel<<<dim3(1024), blk, 0, stream>>>(kfT, ksum);

    // 4) kv^T per (b,h): A=kf^T [256,4096], B^T=v^T [128,4096] -> kv^T [128,256]
    {
        dim3 g(128 / BN, 256 / BM, 32);
        gemm_tdm<OUT_BF16T><<<g, blk, 0, stream>>>(kfT, vT, nullptr, kvT, nullptr,
            4096, 4096, 4096, 256, 1048576LL, 524288LL, 32768LL, 0);
    }

    // 5) denom = q_feat . k_sum
    denom_kernel<<<dim3(16384), blk, 0, stream>>>(qf, ksum, dnm);

    // 6) attn per (b,h): A=qf [4096,256], B^T=kv^T [128,256]; /denom -> ctx
    {
        dim3 g(128 / BN, 4096 / BM, 32);
        gemm_tdm<OUT_CTX><<<g, blk, 0, stream>>>(qf, kvT, nullptr, ctx, dnm,
            256, 256, 256, 0, 1048576LL, 32768LL, 0, 0);
    }

    // 7) output projection: ctx @ Wo + bo -> f32 d_out
    {
        dim3 g(2048 / BN, 8192 / BM, 1);
        gemm_tdm<OUT_F32><<<g, blk, 0, stream>>>(ctx, WoT, bo, d_out, nullptr,
            2048, 2048, 2048, 2048, 0, 0, 0, 0);
    }
}
